// Seq2SeqModel_69020124447177
// MI455X (gfx1250) — matrix operand order repaired and checked by reading, not compiled, not run on hardware
//
#include <hip/hip_runtime.h>
#include <cstdint>
#include <cstddef>

// ---------------- problem constants ----------------
#define BB    256
#define TIN   256
#define TOUT  32
#define EE    512
#define HH    1024
#define YY    1024
#define H3    (3 * HH)

// ---------------- WMMA vector types ----------------
typedef __attribute__((ext_vector_type(16))) __bf16 v16bf;
typedef __attribute__((ext_vector_type(8)))  __bf16 v8bf;
typedef __attribute__((ext_vector_type(8)))  float  v8f;

// ---------------- bf16 helpers ----------------
__device__ __forceinline__ float bf2f(unsigned short u) {
  return __uint_as_float(((unsigned)u) << 16);
}
__device__ __forceinline__ unsigned short f2bf(float f) {
  unsigned u = __float_as_uint(f);
  u += 0x7FFFu + ((u >> 16) & 1u);   // round-to-nearest-even
  return (unsigned short)(u >> 16);
}

// =====================================================================
// WMMA GEMM:  C[M,N] = A[M,K] * W[N,K]^T (+ bias[N])
// A, W row-major bf16; C f32.
// 256 threads = 8 wave32; each wave owns a 16(M) x 64(N) C tile
// (4 accumulators). Per K-step: 1 A fragment + 4 B fragments feed 4
// independent v_wmma_f32_16x16x32_bf16.  Software double-buffering:
// K-step k+1 fragments are fetched into a second register set while
// step k's 4 WMMAs execute, so global-load latency is hidden behind a
// full 4-WMMA window instead of a single matrix op.
// Fragment layouts used here (A and B share one k order):
//   A lane (m = lane&15): K contiguous [h0, h0+16), h0=16*(lane>>4)
//   B lane (n = lane&15): K contiguous [h0, h0+16), h0=16*(lane>>4)
//   D vgpr r: lanes 0-15 -> M=r, N=lane; lanes 16-31 -> M=8+r, N=lane-16
// Requires N % 64 == 0, M % 16 == 0, K % 64 == 0 (true for all launches).
// =====================================================================
__global__ void __launch_bounds__(256)
k_gemm_bf16(const unsigned short* __restrict__ A,
            const unsigned short* __restrict__ W,
            const float* __restrict__ bias,
            float* __restrict__ C, int M, int N, int K)
{
  const int wave   = threadIdx.x >> 5;
  const int lane   = threadIdx.x & 31;
  const int mtile  = blockIdx.y;
  const int ntile0 = (blockIdx.x * 8 + wave) * 4;    // 4 consecutive 16-wide N tiles
  if (ntile0 * 16 >= N) return;                      // wave-uniform exit (EXEC all-1)

  const int lrow = lane & 15;
  const int half = lane >> 4;

  const __bf16* a  = (const __bf16*)A + (size_t)(mtile * 16 + lrow) * K + (half << 4);
  const __bf16* w0 = (const __bf16*)W + (size_t)(ntile0 * 16 + lrow) * K + (half << 4);
  const size_t wstep = (size_t)16 * K;               // next 16-row group of W

  auto loadA = [&](int k) -> v16bf {
    union { v16bf v; v8bf h[2]; } u;
    u.h[0] = *(const v8bf*)(a + k);                  // K = h0 .. h0+7
    u.h[1] = *(const v8bf*)(a + k + 8);              // K = h0+8 .. h0+15
    return u.v;
  };

  v8f acc[4] = {};

  // prologue: fragments for k = 0
  v16bf af  = loadA(0);
  v16bf wf0 = *(const v16bf*)(w0);
  v16bf wf1 = *(const v16bf*)(w0 + wstep);
  v16bf wf2 = *(const v16bf*)(w0 + 2 * wstep);
  v16bf wf3 = *(const v16bf*)(w0 + 3 * wstep);

#pragma unroll 2
  for (int k = 0; k + 32 < K; k += 32) {
    // issue next K-step's loads into the second buffer
    v16bf afn  = loadA(k + 32);
    v16bf wfn0 = *(const v16bf*)(w0 + (k + 32));
    v16bf wfn1 = *(const v16bf*)(w0 + wstep + (k + 32));
    v16bf wfn2 = *(const v16bf*)(w0 + 2 * wstep + (k + 32));
    v16bf wfn3 = *(const v16bf*)(w0 + 3 * wstep + (k + 32));
    // compute current K-step (loads above are covered by these 4 WMMAs)
    acc[0] = __builtin_amdgcn_wmma_f32_16x16x32_bf16(false, af, false, wf0, (short)0, acc[0], false, false);
    acc[1] = __builtin_amdgcn_wmma_f32_16x16x32_bf16(false, af, false, wf1, (short)0, acc[1], false, false);
    acc[2] = __builtin_amdgcn_wmma_f32_16x16x32_bf16(false, af, false, wf2, (short)0, acc[2], false, false);
    acc[3] = __builtin_amdgcn_wmma_f32_16x16x32_bf16(false, af, false, wf3, (short)0, acc[3], false, false);
    // rotate buffers (folds into register renaming under unroll)
    af = afn; wf0 = wfn0; wf1 = wfn1; wf2 = wfn2; wf3 = wfn3;
  }
  // epilogue: last K-step
  acc[0] = __builtin_amdgcn_wmma_f32_16x16x32_bf16(false, af, false, wf0, (short)0, acc[0], false, false);
  acc[1] = __builtin_amdgcn_wmma_f32_16x16x32_bf16(false, af, false, wf1, (short)0, acc[1], false, false);
  acc[2] = __builtin_amdgcn_wmma_f32_16x16x32_bf16(false, af, false, wf2, (short)0, acc[2], false, false);
  acc[3] = __builtin_amdgcn_wmma_f32_16x16x32_bf16(false, af, false, wf3, (short)0, acc[3], false, false);

  const int mbase = mtile * 16 + (half << 3);
#pragma unroll
  for (int i = 0; i < 4; ++i) {
    const int n  = (ntile0 + i) * 16 + lrow;
    const float bv = bias ? bias[n] : 0.0f;
#pragma unroll
    for (int r = 0; r < 8; ++r)
      C[(size_t)(mbase + r) * N + n] = acc[i][r] + bv;
  }
}

// =====================================================================
// GRU gate fusion: h' = (1-z)*n + z*h ; torch gate order (r,z,n)
// Gi/Gh are bias-free GEMM outputs [B,3H]; biases added here.
// Writes f32 hidden + bf16 shadow (+ optional extra bf16 destination).
// =====================================================================
__global__ void __launch_bounds__(256)
k_gru_fuse(const float* __restrict__ Gi, const float* __restrict__ Gh,
           const float* __restrict__ bih, const float* __restrict__ bhh,
           float* __restrict__ h, unsigned short* __restrict__ hb,
           unsigned short* __restrict__ extra, int extra_stride)
{
  int idx = blockIdx.x * blockDim.x + threadIdx.x;
  if (idx >= BB * HH) return;
  int b = idx / HH, j = idx - b * HH;
  const float* gi = Gi + (size_t)b * H3;
  const float* gh = Gh + (size_t)b * H3;
  float ir = gi[j]          + bih[j];
  float iz = gi[HH + j]     + bih[HH + j];
  float in = gi[2 * HH + j] + bih[2 * HH + j];
  float hr = gh[j]          + bhh[j];
  float hz = gh[HH + j]     + bhh[HH + j];
  float hn = gh[2 * HH + j] + bhh[2 * HH + j];
  float r = 1.0f / (1.0f + __expf(-(ir + hr)));
  float z = 1.0f / (1.0f + __expf(-(iz + hz)));
  float n = tanhf(in + r * hn);
  float out = (1.0f - z) * n + z * h[idx];
  h[idx]  = out;
  hb[idx] = f2bf(out);
  if (extra) extra[(size_t)b * extra_stride + j] = f2bf(out);
}

// ---------------- gathers / conversions / zero ----------------
__global__ void k_embed(const int* __restrict__ x, const float* __restrict__ emb,
                        unsigned short* __restrict__ xt, int t)
{
  int idx = blockIdx.x * blockDim.x + threadIdx.x;
  if (idx >= BB * EE) return;
  int b = idx / EE, e = idx - b * EE;
  int tok = x[b * TIN + t];
  xt[idx] = f2bf(emb[(size_t)tok * EE + e]);
}

__global__ void k_yshift(const float* __restrict__ y, unsigned short* __restrict__ yin, int t)
{
  int idx = blockIdx.x * blockDim.x + threadIdx.x;
  if (idx >= BB * YY) return;
  int b = idx / YY, j = idx - b * YY;
  float v = (t == 0) ? 0.0f : y[((size_t)b * TOUT + (t - 1)) * YY + j];
  yin[idx] = f2bf(v);
}

__global__ void k_cvt_bf16(const float* __restrict__ s, unsigned short* __restrict__ d, int n)
{
  for (int i = blockIdx.x * blockDim.x + threadIdx.x; i < n; i += gridDim.x * blockDim.x)
    d[i] = f2bf(s[i]);
}

__global__ void k_zero_bytes(unsigned char* __restrict__ p, size_t n)
{
  size_t stride = (size_t)gridDim.x * blockDim.x;
  for (size_t i = (size_t)blockIdx.x * blockDim.x + threadIdx.x; i < n; i += stride)
    p[i] = 0;
}

// ---------------- attention ----------------
// scores[b,s] = enc[b,s,:] . h2[b,:]   (h2 staged in LDS, one block per b)
__global__ void __launch_bounds__(256)
k_scores(const unsigned short* __restrict__ enc, const unsigned short* __restrict__ h2,
         float* __restrict__ scores)
{
  __shared__ float hs[HH];
  int b = blockIdx.x, tid = threadIdx.x;
  for (int i = tid; i < HH; i += 256) hs[i] = bf2f(h2[(size_t)b * HH + i]);
  __syncthreads();
  const unsigned short* row = enc + ((size_t)b * TIN + tid) * HH;
  float acc = 0.0f;
  for (int i = 0; i < HH; ++i) acc += bf2f(row[i]) * hs[i];
  scores[b * TIN + tid] = acc;
}

// in-place softmax over rows of length TIN (=256 = blockDim)
__global__ void __launch_bounds__(256)
k_softmax(float* __restrict__ s)
{
  __shared__ float red[256];
  int b = blockIdx.x, tid = threadIdx.x;
  float v = s[b * TIN + tid];
  red[tid] = v; __syncthreads();
  for (int o = 128; o > 0; o >>= 1) { if (tid < o) red[tid] = fmaxf(red[tid], red[tid + o]); __syncthreads(); }
  float m = red[0]; __syncthreads();
  float e = __expf(v - m);
  red[tid] = e; __syncthreads();
  for (int o = 128; o > 0; o >>= 1) { if (tid < o) red[tid] += red[tid + o]; __syncthreads(); }
  s[b * TIN + tid] = e / red[0];
}

// attn[b,h] = sum_s alpha[b,s] * enc[b,s,h]; also writes cat[:, :H] in bf16
__global__ void __launch_bounds__(256)
k_attn(const unsigned short* __restrict__ enc, const float* __restrict__ alpha,
       float* __restrict__ attn, unsigned short* __restrict__ cat)
{
  __shared__ float al[TIN];
  int b = blockIdx.x;
  int j = blockIdx.y * 256 + threadIdx.x;
  al[threadIdx.x] = alpha[b * TIN + threadIdx.x];
  __syncthreads();
  const unsigned short* base = enc + (size_t)b * TIN * HH + j;
  float acc = 0.0f;
  for (int sI = 0; sI < TIN; ++sI) acc += al[sI] * bf2f(base[(size_t)sI * HH]);
  attn[(size_t)b * HH + j] = acc;
  cat[(size_t)b * (2 * HH) + j] = f2bf(acc);
}

// ---------------- loss ----------------
// per-row log-softmax over Y=1024, accumulate -(logp*y_t).sum()
__global__ void __launch_bounds__(256)
k_loss(const float* __restrict__ logits, const float* __restrict__ y, int t,
       float* __restrict__ acc)
{
  __shared__ float red[256];
  int b = blockIdx.x, tid = threadIdx.x;
  const float* lr = logits + (size_t)b * YY;
  const float* yr = y + ((size_t)b * TOUT + t) * YY;
  float m = -3.4e38f;
  for (int j = tid; j < YY; j += 256) m = fmaxf(m, lr[j]);
  red[tid] = m; __syncthreads();
  for (int o = 128; o > 0; o >>= 1) { if (tid < o) red[tid] = fmaxf(red[tid], red[tid + o]); __syncthreads(); }
  m = red[0]; __syncthreads();
  float se = 0.0f;
  for (int j = tid; j < YY; j += 256) se += __expf(lr[j] - m);
  red[tid] = se; __syncthreads();
  for (int o = 128; o > 0; o >>= 1) { if (tid < o) red[tid] += red[tid + o]; __syncthreads(); }
  float lse = m + __logf(red[0]); __syncthreads();
  float p = 0.0f;
  for (int j = tid; j < YY; j += 256) p += (lr[j] - lse) * yr[j];
  red[tid] = p; __syncthreads();
  for (int o = 128; o > 0; o >>= 1) { if (tid < o) red[tid] += red[tid + o]; __syncthreads(); }
  if (tid == 0) atomicAdd(acc, -red[0]);
}

__global__ void __launch_bounds__(256)
k_ysum(const float* __restrict__ y, float* __restrict__ dst, int n)
{
  __shared__ float red[256];
  float s = 0.0f;
  for (int i = blockIdx.x * 256 + threadIdx.x; i < n; i += gridDim.x * 256) s += y[i];
  red[threadIdx.x] = s; __syncthreads();
  for (int o = 128; o > 0; o >>= 1) { if (threadIdx.x < o) red[threadIdx.x] += red[threadIdx.x + o]; __syncthreads(); }
  if (threadIdx.x == 0) atomicAdd(dst, red[0]);
}

__global__ void k_fin(const float* __restrict__ s, float* __restrict__ out)
{
  out[0] = s[0] / s[1];
}

// =====================================================================
// host side
// =====================================================================
extern "C" void kernel_launch(void* const* d_in, const int* in_sizes, int n_in,
                              void* d_out, int out_size, void* d_ws, size_t ws_size,
                              hipStream_t stream)
{
  (void)in_sizes; (void)n_in; (void)out_size; (void)ws_size;

  const int*   x    = (const int*)  d_in[0];
  const float* y    = (const float*)d_in[1];
  const float* emb  = (const float*)d_in[2];
  const float* eWih0 = (const float*)d_in[3];
  const float* eWhh0 = (const float*)d_in[4];
  const float* ebih0 = (const float*)d_in[5];
  const float* ebhh0 = (const float*)d_in[6];
  const float* eWih1 = (const float*)d_in[7];
  const float* eWhh1 = (const float*)d_in[8];
  const float* ebih1 = (const float*)d_in[9];
  const float* ebhh1 = (const float*)d_in[10];
  const float* dWih0 = (const float*)d_in[11];
  const float* dWhh0 = (const float*)d_in[12];
  const float* dbih0 = (const float*)d_in[13];
  const float* dbhh0 = (const float*)d_in[14];
  const float* dWih1 = (const float*)d_in[15];
  const float* dWhh1 = (const float*)d_in[16];
  const float* dbih1 = (const float*)d_in[17];
  const float* dbhh1 = (const float*)d_in[18];
  const float* mapW  = (const float*)d_in[19];
  const float* mapb  = (const float*)d_in[20];

  // ---- workspace carve ----
  char* p = (char*)d_ws;
  auto carve = [&](size_t bytes) -> char* {
    char* r = p; p += (bytes + 255) & ~(size_t)255; return r;
  };
  float*          scal  = (float*)carve(2 * sizeof(float));       // [loss_acc, y_sum]
  float*          hAf   = (float*)carve((size_t)BB * HH * 4);     // enc L0 / dec L0 hidden
  float*          hBf   = (float*)carve((size_t)BB * HH * 4);     // enc L1 / dec L1 hidden
  unsigned short* hAb   = (unsigned short*)carve((size_t)BB * HH * 2);
  unsigned short* hBb   = (unsigned short*)carve((size_t)BB * HH * 2);
  size_t zero_span = (size_t)(p - (char*)scal);                   // zero scalars + hiddens

  unsigned short* xtb   = (unsigned short*)carve((size_t)BB * EE * 2);
  unsigned short* yinb  = (unsigned short*)carve((size_t)BB * YY * 2);
  float*          Gi    = (float*)carve((size_t)BB * H3 * 4);
  float*          Gh    = (float*)carve((size_t)BB * H3 * 4);
  float*          alp   = (float*)carve((size_t)BB * TIN * 4);    // scores -> alpha
  float*          attn  = (float*)carve((size_t)BB * HH * 4);
  unsigned short* catb  = (unsigned short*)carve((size_t)BB * 2 * HH * 2);
  float*          logit = (float*)carve((size_t)BB * YY * 4);
  unsigned short* encb  = (unsigned short*)carve((size_t)BB * TIN * HH * 2);

  unsigned short* bWeih0 = (unsigned short*)carve((size_t)H3 * EE * 2);
  unsigned short* bWehh0 = (unsigned short*)carve((size_t)H3 * HH * 2);
  unsigned short* bWeih1 = (unsigned short*)carve((size_t)H3 * HH * 2);
  unsigned short* bWehh1 = (unsigned short*)carve((size_t)H3 * HH * 2);
  unsigned short* bWdih0 = (unsigned short*)carve((size_t)H3 * YY * 2);
  unsigned short* bWdhh0 = (unsigned short*)carve((size_t)H3 * HH * 2);
  unsigned short* bWdih1 = (unsigned short*)carve((size_t)H3 * HH * 2);
  unsigned short* bWdhh1 = (unsigned short*)carve((size_t)H3 * HH * 2);
  unsigned short* bWmap  = (unsigned short*)carve((size_t)YY * 2 * HH * 2);

  auto cvt = [&](const float* s, unsigned short* d, int n) {
    int g = (n + 255) / 256; if (g > 2048) g = 2048;
    k_cvt_bf16<<<g, 256, 0, stream>>>(s, d, n);
  };
  // per-wave tile = 16x64  ->  block covers 512 columns of C
  auto gemm = [&](const unsigned short* A, const unsigned short* W, const float* bias,
                  float* C, int M, int N, int K) {
    dim3 grid((N / 64 + 7) / 8, M / 16);
    k_gemm_bf16<<<grid, 256, 0, stream>>>(A, W, bias, C, M, N, K);
  };

  // ---- init ----
  k_zero_bytes<<<512, 256, 0, stream>>>((unsigned char*)scal, zero_span);
  k_ysum<<<1024, 256, 0, stream>>>(y, scal + 1, BB * TOUT * YY);

  cvt(eWih0, bWeih0, H3 * EE);
  cvt(eWhh0, bWehh0, H3 * HH);
  cvt(eWih1, bWeih1, H3 * HH);
  cvt(eWhh1, bWehh1, H3 * HH);
  cvt(dWih0, bWdih0, H3 * YY);
  cvt(dWhh0, bWdhh0, H3 * HH);
  cvt(dWih1, bWdih1, H3 * HH);
  cvt(dWhh1, bWdhh1, H3 * HH);
  cvt(mapW,  bWmap,  YY * 2 * HH);

  // ---- encoder: both layers advanced per time step ----
  for (int t = 0; t < TIN; ++t) {
    k_embed<<<(BB * EE) / 256, 256, 0, stream>>>(x, emb, xtb, t);
    // layer 0
    gemm(xtb, bWeih0, nullptr, Gi, BB, H3, EE);
    gemm(hAb, bWehh0, nullptr, Gh, BB, H3, HH);
    k_gru_fuse<<<(BB * HH) / 256, 256, 0, stream>>>(Gi, Gh, ebih0, ebhh0,
                                                    hAf, hAb, nullptr, 0);
    // layer 1 (output kept as enc_out[b, t, :] in bf16 for attention)
    gemm(hAb, bWeih1, nullptr, Gi, BB, H3, HH);
    gemm(hBb, bWehh1, nullptr, Gh, BB, H3, HH);
    k_gru_fuse<<<(BB * HH) / 256, 256, 0, stream>>>(Gi, Gh, ebih1, ebhh1,
                                                    hBf, hBb,
                                                    encb + (size_t)t * HH, TIN * HH);
  }

  // ---- decoder: hA/hB carry the encoder finals as initial hidden ----
  for (int t = 0; t < TOUT; ++t) {
    k_yshift<<<(BB * YY) / 256, 256, 0, stream>>>(y, yinb, t);
    // layer 0
    gemm(yinb, bWdih0, nullptr, Gi, BB, H3, YY);
    gemm(hAb,  bWdhh0, nullptr, Gh, BB, H3, HH);
    k_gru_fuse<<<(BB * HH) / 256, 256, 0, stream>>>(Gi, Gh, dbih0, dbhh0,
                                                    hAf, hAb, nullptr, 0);
    // layer 1; bf16 copy of h2 also lands in cat[:, H:2H]
    gemm(hAb, bWdih1, nullptr, Gi, BB, H3, HH);
    gemm(hBb, bWdhh1, nullptr, Gh, BB, H3, HH);
    k_gru_fuse<<<(BB * HH) / 256, 256, 0, stream>>>(Gi, Gh, dbih1, dbhh1,
                                                    hBf, hBb, catb + HH, 2 * HH);
    // attention over encoder positions
    k_scores<<<BB, 256, 0, stream>>>(encb, hBb, alp);
    k_softmax<<<BB, 256, 0, stream>>>(alp);
    k_attn<<<dim3(BB, HH / 256), 256, 0, stream>>>(encb, alp, attn, catb);
    // logits = [attn, h2] @ mapW^T + mapb ; then loss accumulation
    gemm(catb, bWmap, mapb, logit, BB, YY, 2 * HH);
    k_loss<<<BB, 256, 0, stream>>>(logit, y, t, scal);
  }

  k_fin<<<1, 1, 0, stream>>>(scal, (float*)d_out);
}
